// Block_65859028516968
// MI455X (gfx1250) — matrix operand order repaired and checked by reading, not compiled, not run on hardware
//
#include <hip/hip_runtime.h>
#include <math.h>

// ---------------- problem constants ----------------
#define B_   2
#define NN_  4096
#define E_   16384
#define EB_  8192
constexpr int   BN  = B_ * NN_;     // 8192 node rows
constexpr int   MV  = BN * 3;       // 24576 vector rows (v3 layout)
constexpr float R0c = 6.0f;

typedef __attribute__((ext_vector_type(16))) _Float16 v16h;
typedef __attribute__((ext_vector_type(8)))  float    v8f;

#define GF_LEAKY 1
#define GF_BIAS  2
#define GF_ACC   4

// =====================================================================
// Fused WMMA GEMM:  D[M,N] = act(A[M,K] @ W[K,N] + bias) (+= D if ACC)
// A,W,D row-major fp32. K in {64,128}, N in {64,128}, M % 64 == 0.
// Block = 128 threads (4 waves). Wave w computes rows [row0, row0+16).
// W tile (K x 64 fp32) is DMA'd into LDS with GLOBAL_LOAD_ASYNC_TO_LDS_B128
// (ASYNCcnt), then swizzled in LDS into the CDNA5 B-matrix fragment layout
// so each lane reads its v16h fragment as two contiguous ds_load_b128.
// =====================================================================
__global__ void __launch_bounds__(128)
gemm_wmma(const float* __restrict__ A, const float* __restrict__ W,
          const float* __restrict__ bias, float* __restrict__ D,
          int M, int K, int N, int flags)
{
    const int lane   = threadIdx.x & 31;
    const int wave   = threadIdx.x >> 5;
    const int row0   = blockIdx.x * 64 + wave * 16;
    const int col0   = blockIdx.y * 64;
    const int halfid = lane >> 4;      // lane group (0: lanes 0-15, 1: 16-31)
    const int m      = lane & 15;

    // raw async-staged W tile (max 128x64 f32 = 32KB)
    __shared__ __align__(16) float Wraw[128 * 64];
    // [kchunk][coltile][lane][16 halves]  (max 128x64 f16 = 16KB)
    __shared__ __align__(32) _Float16 Ws[4][4][32][16];

    const int nK    = K >> 5;
    const int total = K * 64;          // tile elements (fp32)

    // ---- stage 1: async DMA global -> LDS, 16B per lane-op ----
    for (int idx4 = threadIdx.x * 4; idx4 < total; idx4 += 128 * 4) {
        int k  = idx4 >> 6;
        int nn = idx4 & 63;
        unsigned long long gaddr =
            (unsigned long long)(W + (size_t)k * N + col0 + nn);
        unsigned lds_off = (unsigned)(uintptr_t)
            ((__attribute__((address_space(3))) float*)&Wraw[idx4]);
        asm volatile("global_load_async_to_lds_b128 %0, %1, off"
                     :: "v"(lds_off), "v"(gaddr) : "memory");
    }
    asm volatile("s_wait_asynccnt 0" ::: "memory");
    __syncthreads();

    // ---- stage 2: swizzle LDS fp32 -> LDS f16 B-fragment layout ----
    for (int idx = threadIdx.x; idx < total; idx += 128) {
        int k  = idx >> 6;
        int nn = idx & 63;
        float w = Wraw[idx];
        int kc = k >> 5, kk = k & 31;
        int ct = nn >> 4;
        int ln = (((kk >> 3) & 1) << 4) | (nn & 15);   // bit 3 of K selects lane group
        Ws[kc][ct][ln][((kk >> 4) << 3) | (kk & 7)] = (_Float16)w; // elem s: K=8h+s (s<8), 16+8h+s-8 (s>=8)
    }
    __syncthreads();

    v8f acc[4];
#pragma unroll
    for (int ct = 0; ct < 4; ++ct)
#pragma unroll
        for (int i = 0; i < 8; ++i) acc[ct][i] = 0.0f;

    const float* arow = A + (size_t)(row0 + m) * K;
    for (int kc = 0; kc < nK; ++kc) {
        if (kc + 1 < nK) __builtin_prefetch(arow + (kc + 1) * 32, 0, 3);
        // A fragment (16-bit A 16x32 layout): lane needs K = kc*32 + h*8 + 0..7
        // in elements 0..7, and +16 in elements 8..15.
        const float* p0 = arow + kc * 32 + halfid * 8;
        float a8[8], b8[8];
        *(float4*)(a8)     = ((const float4*)p0)[0];
        *(float4*)(a8 + 4) = ((const float4*)p0)[1];
        *(float4*)(b8)     = ((const float4*)(p0 + 16))[0];
        *(float4*)(b8 + 4) = ((const float4*)(p0 + 16))[1];
        v16h afrag;
#pragma unroll
        for (int t = 0; t < 8; ++t) { afrag[t] = (_Float16)a8[t]; afrag[8 + t] = (_Float16)b8[t]; }
#pragma unroll
        for (int ct = 0; ct < 4; ++ct) {
            v16h bfrag = *(const v16h*)(&Ws[kc][ct][lane][0]);
            acc[ct] = __builtin_amdgcn_wmma_f32_16x16x32_f16(
                false, afrag, false, bfrag, (short)0, acc[ct], false, false);
        }
    }

    // ---- epilogue: C/D layout -> lane n = col, VGPR v -> row (+8 for hi half)
#pragma unroll
    for (int ct = 0; ct < 4; ++ct) {
        int n = col0 + ct * 16 + m;
        float bv = (flags & GF_BIAS) ? bias[n] : 0.0f;
#pragma unroll
        for (int v = 0; v < 8; ++v) {
            int row = row0 + halfid * 8 + v;
            float y = acc[ct][v] + bv;
            if (flags & GF_LEAKY) y = (y > 0.0f) ? y : 0.1f * y;
            float* dp = D + (size_t)row * N + n;
            if (flags & GF_ACC) y += *dp;
            *dp = y;
        }
    }
}

// =====================================================================
// Elementwise / gather / scatter kernels
// =====================================================================
__global__ void k_temb(const float* __restrict__ t, const float* __restrict__ w,
                       const float* __restrict__ bias, float* __restrict__ out, int outdim)
{
    int b = blockIdx.x, o = threadIdx.x;
    if (o >= outdim) return;
    float tv = t[b], acc = bias[o];
    for (int i = 0; i < 32; ++i) {
        float ph = tv * __expf(-(float)i / 31.0f);
        acc += __sinf(ph) * w[i * outdim + o] + __cosf(ph) * w[(32 + i) * outdim + o];
    }
    out[b * outdim + o] = acc;
}

__global__ void k_gather_res(const int* __restrict__ md, const float* __restrict__ emb,
                             float* __restrict__ resg)
{
    int bn = blockIdx.x, c = threadIdx.x;
    resg[(size_t)bn * 128 + c] = emb[md[bn] * 128 + c];
}

__global__ void k_edge_rvec(const float* __restrict__ pos, const int* __restrict__ src,
                            const int* __restrict__ dst, float* __restrict__ r, int Ecnt)
{
    int i = blockIdx.x * blockDim.x + threadIdx.x;
    if (i >= B_ * Ecnt * 3) return;
    int x = i % 3, row = i / 3, e = row % Ecnt, b = row / Ecnt;
    r[i] = pos[((size_t)(b * NN_ + dst[e])) * 3 + x] - pos[((size_t)(b * NN_ + src[e])) * 3 + x];
}

// radial encode (+ optional x_a[dst]+res[dst]), and rs = tens_sigmoid(r*7/R0)
__global__ void k_edge_enc(const float* __restrict__ r, const int* __restrict__ dst,
                           const float* __restrict__ xaadd, const float* __restrict__ resg,
                           float* __restrict__ enc, float* __restrict__ rs, int Ecnt)
{
    int row = blockIdx.x, c = threadIdx.x;
    int e = row % Ecnt, b = row / Ecnt;
    float r0v = r[(size_t)row * 3 + 0], r1v = r[(size_t)row * 3 + 1], r2v = r[(size_t)row * 3 + 2];
    float sq  = (r0v * r0v + r1v * r1v + r2v * r2v) / (R0c * R0c);
    float hyp = sqrtf(1.0f + sq);
    float coef = 2.2214414691f * __expf(0.0334929225f * (float)c); // pi/sqrt2 * exp(128^-0.7 * c)
    float val = __sinf(coef * hyp) * (0.70710678f - hyp);
    if (xaadd) {
        size_t nb = (size_t)(b * NN_ + dst[e]) * 128 + c;
        val += xaadd[nb] + resg[nb];
    }
    enc[(size_t)row * 128 + c] = val;
    if (c < 3) {
        float s7 = 7.0f / R0c;
        float v0 = r0v * s7, v1 = r1v * s7, v2 = r2v * s7;
        float inv = rsqrtf(1.0f + v0 * v0 + v1 * v1 + v2 * v2);
        rs[(size_t)row * 3 + c] = (c == 0 ? v0 : (c == 1 ? v1 : v2)) * inv;
    }
}

__global__ void k_scatter_a(const float* __restrict__ val, const int* __restrict__ src,
                            float* __restrict__ acc, float scale, int Ecnt)
{
    int row = blockIdx.x, c = threadIdx.x, e = row % Ecnt, b = row / Ecnt;
    atomicAdd(&acc[(size_t)(b * NN_ + src[e]) * 128 + c], scale * val[(size_t)row * 128 + c]);
}

__global__ void k_scatter_pv(const float* __restrict__ rad, const float* __restrict__ rs,
                             const int* __restrict__ src, float* __restrict__ Pv, int Ecnt)
{
    int row = blockIdx.x, c = threadIdx.x, e = row % Ecnt, b = row / Ecnt;
    float rv = rad[(size_t)row * 128 + c];
    size_t base = (size_t)(b * NN_ + src[e]) * 3;
    for (int x = 0; x < 3; ++x)
        atomicAdd(&Pv[(base + x) * 128 + c], rv * rs[(size_t)row * 3 + x]);
}

__global__ void k_scatter_v64(const float* __restrict__ val, const int* __restrict__ src,
                              float* __restrict__ accv3, float scale, int comp, int Ecnt)
{
    int row = blockIdx.x, c = threadIdx.x, e = row % Ecnt, b = row / Ecnt;
    atomicAdd(&accv3[((size_t)(b * NN_ + src[e]) * 3 + comp) * 64 + c],
              scale * val[(size_t)row * 64 + c]);
}

__global__ void k_build_xa1(const float* xa_in, const float* acc_a, const float* resg,
                            const float* te, float* xa1)
{
    int bn = blockIdx.x, c = threadIdx.x, b = bn / NN_;
    size_t i = (size_t)bn * 128 + c;
    xa1[i] = xa_in[i] + acc_a[i] + resg[i] + te[b * 128 + c];
}

__global__ void k_xv_to_v3(const float* xv_in, float* v3)
{
    int bn = blockIdx.x, c = threadIdx.x;
    for (int x = 0; x < 3; ++x)
        v3[((size_t)bn * 3 + x) * 64 + c] = xv_in[((size_t)bn * 64 + c) * 3 + x];
}

__global__ void k_build_xv1(const float* xvin_v3, const float* acc_v, const float* pos0,
                            const float* pos1, const float* posW, float* xv1)
{
    int bn = blockIdx.x, c = threadIdx.x;
    for (int x = 0; x < 3; ++x) {
        float dp = pos1[(size_t)bn * 3 + x] - pos0[(size_t)bn * 3 + x];
        size_t i = ((size_t)bn * 3 + x) * 64 + c;
        xv1[i] = xvin_v3[i] + acc_v[i] + dp * posW[c];
    }
}

__global__ void k_mul(float* out, const float* a, const float* b, int n)
{
    int i = blockIdx.x * blockDim.x + threadIdx.x;
    if (i < n) out[i] = a[i] * b[i];
}
__global__ void k_add(float* out, const float* a, const float* b, int n)
{
    int i = blockIdx.x * blockDim.x + threadIdx.x;
    if (i < n) out[i] = a[i] + b[i];
}

__global__ void k_dot3(const float* l1, const float* r1, float* out)
{
    int bn = blockIdx.x, c = threadIdx.x;
    float s = 0.0f;
    for (int x = 0; x < 3; ++x) {
        size_t i = ((size_t)bn * 3 + x) * 128 + c;
        s += l1[i] * r1[i];
    }
    out[(size_t)bn * 128 + c] = s;
}

__global__ void k_nodevec_mul(const float* sn, const float* vec, float* u)
{
    int row = blockIdx.x, c = threadIdx.x, bn = row / 3;
    u[(size_t)row * 128 + c] = sn[(size_t)bn * 128 + c] * vec[(size_t)row * 128 + c];
}

__global__ void k_cross_node(const float* l1, const float* r1, float* u)
{
    int row = blockIdx.x, c = threadIdx.x, bn = row / 3, x = row % 3;
    int y = (x + 1) % 3, z = (x + 2) % 3;
    size_t iy = ((size_t)bn * 3 + y) * 128 + c, iz = ((size_t)bn * 3 + z) * 128 + c;
    u[(size_t)row * 128 + c] = l1[iy] * r1[iz] - l1[iz] * r1[iy];
}

__global__ void k_gn0(const float* x, const float* gamma, float* out)
{
    __shared__ float sh[128];
    int bn = blockIdx.x, c = threadIdx.x;
    float v = x[(size_t)bn * 128 + c];
    sh[c] = v;
    __syncthreads();
    int g = c >> 4;
    float mu = 0.0f;
    for (int j = 0; j < 16; ++j) mu += sh[g * 16 + j];
    mu *= (1.0f / 16.0f);
    float var = 0.0f;
    for (int j = 0; j < 16; ++j) { float d = sh[g * 16 + j] - mu; var += d * d; }
    var *= (1.0f / 16.0f);
    out[(size_t)bn * 128 + c] = (v - mu) * rsqrtf(var + 1e-5f) * gamma[c];
}

__global__ void k_gn1(float* x, const float* gamma)
{
    __shared__ float sh[64];
    int bn = blockIdx.x, c = threadIdx.x;
    size_t i0 = ((size_t)bn * 3 + 0) * 64 + c, i1 = i0 + 64, i2 = i0 + 128;
    float v0 = x[i0], v1 = x[i1], v2 = x[i2];
    sh[c] = v0 * v0 + v1 * v1 + v2 * v2;
    __syncthreads();
    int g = c >> 3;
    float ms = 0.0f;
    for (int j = 0; j < 8; ++j) ms += sh[g * 8 + j];
    ms *= (1.0f / 24.0f);
    float s = rsqrtf(ms + 1e-5f) * gamma[c];
    x[i0] = v0 * s; x[i1] = v1 * s; x[i2] = v2 * s;
}

__global__ void k_msg_terms(const float* rad, const float* rs, const int* dst,
                            const float* A0, const float* V1, float* pa, float* p110, int Ecnt)
{
    int row = blockIdx.x, c = threadIdx.x, e = row % Ecnt, b = row / Ecnt;
    int d = dst[e];
    float rv = rad[(size_t)row * 128 + c];
    pa[(size_t)row * 128 + c] = A0[(size_t)(b * NN_ + d) * 128 + c] * rv;
    float s = 0.0f;
    for (int x = 0; x < 3; ++x)
        s += V1[((size_t)(b * NN_ + d) * 3 + x) * 128 + c] * rs[(size_t)row * 3 + x];
    p110[(size_t)row * 128 + c] = rv * s;
}

__global__ void k_msg_q(const float* rad, const float* rs, const int* dst,
                        const float* A0, const float* V1, float* q, int mode, int comp, int Ecnt)
{
    int row = blockIdx.x, c = threadIdx.x, e = row % Ecnt, b = row / Ecnt;
    int d = dst[e];
    float rv = rad[(size_t)row * 128 + c];
    size_t nb = (size_t)(b * NN_ + d) * 3;
    float o;
    if (mode == 0)      o = A0[(size_t)(b * NN_ + d) * 128 + c] * rv * rs[(size_t)row * 3 + comp];
    else if (mode == 1) o = V1[(nb + comp) * 128 + c] * rv;
    else {
        int y = (comp + 1) % 3, z = (comp + 2) % 3;
        o = rv * (V1[(nb + y) * 128 + c] * rs[(size_t)row * 3 + z]
                - V1[(nb + z) * 128 + c] * rs[(size_t)row * 3 + y]);
    }
    q[(size_t)row * 128 + c] = o;
}

__global__ void k_scale_bcast(float* h, const float* emb, int rowsPerB, int C, int total)
{
    int i = blockIdx.x * blockDim.x + threadIdx.x;
    if (i >= total) return;
    int c = i % C, b = (i / C) / rowsPerB;
    h[i] *= emb[b * C + c];
}

__global__ void k_out(const float* xa, const float* xv3, float* out)
{
    int bn = blockIdx.x, c = threadIdx.x;
    out[(size_t)bn * 128 + c] = xa[(size_t)bn * 128 + c];
    if (c < 64)
        for (int x = 0; x < 3; ++x)
            out[(size_t)BN * 128 + ((size_t)bn * 64 + c) * 3 + x] =
                xv3[((size_t)bn * 3 + x) * 64 + c];
}

// =====================================================================
// Host-side parameter offsets + orchestration
// =====================================================================
struct MlpOff { size_t w1, b1, w2, b2, w3; };
struct DtOff  { MlpOff mlp; size_t ro; };
struct StpOff { size_t L0, R0, L1, R1, updl, updr, Y000, Y110, Y011, Y101, Y111; MlpOff mlp; size_t gna, gnv; };
struct MsgOff { size_t L0, L1, Y000, Y110, Y011, Y101, Y111, encw, encb; MlpOff mlp; };

static inline void gemm(hipStream_t s, const float* A, const float* W, const float* bias,
                        float* D, int M, int K, int N, int flags)
{
    dim3 g(M / 64, N / 64);
    gemm_wmma<<<g, 128, 0, s>>>(A, W, bias, D, M, K, N, flags);
}

static void run_disp(hipStream_t s, const float* P, const DtOff& po,
                     const float* r, const int* src, const int* dst,
                     const float* xa_in, const float* resg,
                     float* acc_a, float* Pv, float* enc, float* rs, float* h1, float* h2)
{
    const int Me = B_ * EB_;
    k_edge_enc<<<Me, 128, 0, s>>>(r, dst, xa_in, resg, enc, rs, EB_);
    gemm(s, enc, P + po.mlp.w1, P + po.mlp.b1, h1, Me, 128, 128, GF_BIAS | GF_LEAKY);
    gemm(s, h1, P + po.mlp.w2, P + po.mlp.b2, h2, Me, 128, 128, GF_BIAS | GF_LEAKY);
    gemm(s, h2, P + po.mlp.w3, nullptr, enc, Me, 128, 128, 0);   // enc <- rad
    k_scatter_a<<<Me, 128, 0, s>>>(enc, src, acc_a, 1.0f, EB_);
    k_scatter_pv<<<Me, 128, 0, s>>>(enc, rs, src, Pv, EB_);
}

static void run_stp(hipStream_t s, const float* P, const StpOff& po,
                    const float* xa1, const float* xv1,
                    float* l0, float* r0, float* l1, float* r1, float* ya,
                    float* tA, float* tB, float* u, float* da, float* dv)
{
    gemm(s, xa1, P + po.L0, nullptr, l0, BN, 128, 128, 0);
    gemm(s, xa1, P + po.R0, nullptr, r0, BN, 128, 128, 0);
    gemm(s, xv1, P + po.L1, nullptr, l1, MV, 64, 128, 0);
    gemm(s, xv1, P + po.R1, nullptr, r1, MV, 64, 128, 0);
    k_mul<<<(BN * 128) / 256, 256, 0, s>>>(tA, l0, r0, BN * 128);
    gemm(s, tA, P + po.Y000, nullptr, ya, BN, 128, 128, 0);
    k_dot3<<<BN, 128, 0, s>>>(l1, r1, tA);
    gemm(s, tA, P + po.Y110, nullptr, ya, BN, 128, 128, GF_ACC);
    gemm(s, ya, P + po.mlp.w1, P + po.mlp.b1, tA, BN, 128, 128, GF_BIAS | GF_LEAKY);
    gemm(s, tA, P + po.mlp.w2, P + po.mlp.b2, tB, BN, 128, 128, GF_BIAS | GF_LEAKY);
    gemm(s, tB, P + po.mlp.w3, nullptr, ya, BN, 128, 128, GF_ACC);
    gemm(s, ya, P + po.updl, nullptr, l0, BN, 128, 128, GF_ACC);
    gemm(s, ya, P + po.updr, nullptr, r0, BN, 128, 128, GF_ACC);
    k_nodevec_mul<<<MV, 128, 0, s>>>(l0, r1, u);
    gemm(s, u, P + po.Y011, nullptr, dv, MV, 128, 64, 0);
    k_nodevec_mul<<<MV, 128, 0, s>>>(r0, l1, u);
    gemm(s, u, P + po.Y101, nullptr, dv, MV, 128, 64, GF_ACC);
    k_cross_node<<<MV, 128, 0, s>>>(l1, r1, u);
    gemm(s, u, P + po.Y111, nullptr, dv, MV, 128, 64, GF_ACC);
    k_gn0<<<BN, 128, 0, s>>>(ya, P + po.gna, da);
    k_gn1<<<BN, 64, 0, s>>>(dv, P + po.gnv);
}

static void run_msg(hipStream_t s, const float* P, const MsgOff& po,
                    const float* r, const int* src, const int* dst, int Ecnt,
                    const float* xa, const float* xv3, float* xa_acc, float* xv_acc,
                    float* A0, float* V1, float* enc, float* rs, float* rad,
                    float* t1, float* t2, float* psia, float* q, float* psiv)
{
    const int Me = B_ * Ecnt;
    k_edge_enc<<<Me, 128, 0, s>>>(r, dst, nullptr, nullptr, enc, rs, Ecnt);
    gemm(s, enc, P + po.encw, P + po.encb, rad, Me, 128, 128, GF_BIAS);
    gemm(s, xa, P + po.L0, nullptr, A0, BN, 128, 128, 0);
    gemm(s, xv3, P + po.L1, nullptr, V1, MV, 64, 128, 0);
    k_msg_terms<<<Me, 128, 0, s>>>(rad, rs, dst, A0, V1, t1, t2, Ecnt);
    gemm(s, t1, P + po.Y000, nullptr, psia, Me, 128, 128, 0);
    gemm(s, t2, P + po.Y110, nullptr, psia, Me, 128, 128, GF_ACC);
    gemm(s, psia, P + po.mlp.w1, P + po.mlp.b1, t1, Me, 128, 128, GF_BIAS | GF_LEAKY);
    gemm(s, t1, P + po.mlp.w2, P + po.mlp.b2, t2, Me, 128, 128, GF_BIAS | GF_LEAKY);
    gemm(s, t2, P + po.mlp.w3, nullptr, psia, Me, 128, 128, GF_ACC);
    k_scatter_a<<<Me, 128, 0, s>>>(psia, src, xa_acc, 0.1f, Ecnt);
    for (int x = 0; x < 3; ++x) {
        k_msg_q<<<Me, 128, 0, s>>>(rad, rs, dst, A0, V1, q, 0, x, Ecnt);
        gemm(s, q, P + po.Y011, nullptr, psiv, Me, 128, 64, 0);
        k_msg_q<<<Me, 128, 0, s>>>(rad, rs, dst, A0, V1, q, 1, x, Ecnt);
        gemm(s, q, P + po.Y101, nullptr, psiv, Me, 128, 64, GF_ACC);
        k_msg_q<<<Me, 128, 0, s>>>(rad, rs, dst, A0, V1, q, 2, x, Ecnt);
        gemm(s, q, P + po.Y111, nullptr, psiv, Me, 128, 64, GF_ACC);
        k_scatter_v64<<<Me, 64, 0, s>>>(psiv, src, xv_acc, 0.1f, x, Ecnt);
    }
}

extern "C" void kernel_launch(void* const* d_in, const int* in_sizes, int n_in,
                              void* d_out, int out_size, void* d_ws, size_t ws_size,
                              hipStream_t stream)
{
    (void)in_sizes; (void)n_in; (void)out_size; (void)ws_size;
    const float* t     = (const float*)d_in[0];
    const float* xa_in = (const float*)d_in[1];
    const float* xv_in = (const float*)d_in[2];
    const float* pos0  = (const float*)d_in[3];
    const float* pos1  = (const float*)d_in[4];
    const float* rij0  = (const float*)d_in[5];
    const float* rij1  = (const float*)d_in[6];
    const int*   meta  = (const int*)d_in[7];
    const int*   src0  = (const int*)d_in[8];
    const int*   dst0  = (const int*)d_in[9];
    const int*   src1  = (const int*)d_in[10];
    const int*   dst1  = (const int*)d_in[11];
    const int*   srcb  = (const int*)d_in[12];
    const int*   dstb  = (const int*)d_in[13];
    const float* P     = (const float*)d_in[14];
    float*       out   = (float*)d_out;

    // ---- parameter offsets (make_params literal order, depth-first) ----
    size_t o = 0;
    auto L = [&](size_t n) { size_t r = o; o += n; return r; };
    auto mlpo = [&](MlpOff& m) { m.w1 = L(16384); m.b1 = L(128); m.w2 = L(16384); m.b2 = L(128); m.w3 = L(16384); };
    size_t p_res = L(32 * 128);
    DtOff dt0, dt1; mlpo(dt0.mlp); dt0.ro = L(128 * 64); mlpo(dt1.mlp); dt1.ro = L(128 * 64);
    auto stpo = [&](StpOff& sp) {
        sp.L0 = L(16384); sp.R0 = L(16384); sp.L1 = L(8192); sp.R1 = L(8192);
        sp.updl = L(16384); sp.updr = L(16384); sp.Y000 = L(16384); sp.Y110 = L(16384);
        sp.Y011 = L(8192); sp.Y101 = L(8192); sp.Y111 = L(8192); mlpo(sp.mlp);
        sp.gna = L(128); sp.gnv = L(64);
    };
    StpOff stpA, stpB; stpo(stpA); stpo(stpB);
    auto msgo = [&](MsgOff& m) {
        m.L0 = L(16384); m.L1 = L(8192); m.Y000 = L(16384); m.Y110 = L(16384);
        m.Y011 = L(8192); m.Y101 = L(8192); m.Y111 = L(8192); m.encw = L(16384); m.encb = L(128); mlpo(m.mlp);
    };
    MsgOff m0, m1, mb0, mb1; msgo(m0); msgo(m1); msgo(mb0); msgo(mb1);
    size_t p_et_w = L(64 * 128), p_et_b = L(128);
    size_t p_ta_ew = L(64 * 128), p_ta_eb = L(128), p_ta_i = L(16384), p_ta_o = L(16384);
    size_t p_tv_ew = L(64 * 64), p_tv_eb = L(64), p_tv_i = L(4096), p_tv_o = L(4096);
    size_t p_pos = L(64);

    // ---- workspace bump allocator ----
    float* base = (float*)d_ws;
    size_t off = 0;
    auto alloc = [&](size_t n) { float* p = base + off; off += ((n + 15) & ~(size_t)15); return p; };
    float* resg    = alloc((size_t)BN * 128);
    float* te_a    = alloc(B_ * 128);
    float* tlmte_a = alloc(B_ * 128);
    float* tlmte_v = alloc(B_ * 64);
    float* xvin_v3 = alloc((size_t)MV * 64);
    float* xa_cur  = alloc((size_t)BN * 128);
    float* xv_cur  = alloc((size_t)MV * 64);
    float* acc_a   = alloc((size_t)BN * 128);
    float* acc_v   = alloc((size_t)MV * 64);
    float* rb0     = alloc((size_t)B_ * EB_ * 3);
    float* rb1     = alloc((size_t)B_ * EB_ * 3);
    float* arena   = base + off;
    size_t aoff;
    auto salloc = [&](size_t n) { float* p = arena + aoff; aoff += ((n + 15) & ~(size_t)15); return p; };

    // ================= stage 0: embeddings, layout conversion =================
    k_gather_res<<<BN, 128, 0, stream>>>(meta, P + p_res, resg);
    k_temb<<<B_, 128, 0, stream>>>(t, P + p_et_w, P + p_et_b, te_a, 128);
    k_temb<<<B_, 128, 0, stream>>>(t, P + p_ta_ew, P + p_ta_eb, tlmte_a, 128);
    k_temb<<<B_, 64, 0, stream>>>(t, P + p_tv_ew, P + p_tv_eb, tlmte_v, 64);
    k_xv_to_v3<<<BN, 64, 0, stream>>>(xv_in, xvin_v3);
    k_edge_rvec<<<(B_ * EB_ * 3 + 255) / 256, 256, 0, stream>>>(pos0, srcb, dstb, rb0, EB_);
    k_edge_rvec<<<(B_ * EB_ * 3 + 255) / 256, 256, 0, stream>>>(pos1, srcb, dstb, rb1, EB_);

    // ================= stage 1: disp_tens (dt0, dt1) =================
    aoff = 0;
    {
        const size_t Me = (size_t)B_ * EB_;
        float* enc = salloc(Me * 128);
        float* h1  = salloc(Me * 128);
        float* h2  = salloc(Me * 128);
        float* rs  = salloc(Me * 3);
        float* Pv  = salloc((size_t)MV * 128);
        hipMemsetAsync(acc_a, 0, (size_t)BN * 128 * 4, stream);
        hipMemsetAsync(Pv, 0, (size_t)MV * 128 * 4, stream);
        run_disp(stream, P, dt0, rb0, srcb, dstb, xa_in, resg, acc_a, Pv, enc, rs, h1, h2);
        gemm(stream, Pv, P + dt0.ro, nullptr, acc_v, MV, 128, 64, 0);
        hipMemsetAsync(Pv, 0, (size_t)MV * 128 * 4, stream);
        run_disp(stream, P, dt1, rb1, srcb, dstb, xa_in, resg, acc_a, Pv, enc, rs, h1, h2);
        gemm(stream, Pv, P + dt1.ro, nullptr, acc_v, MV, 128, 64, GF_ACC);
    }

    // ================= stage 2: self_tens_prods A =================
    aoff = 0;
    {
        float* xa1 = salloc((size_t)BN * 128); float* xv1 = salloc((size_t)MV * 64);
        float* l0 = salloc((size_t)BN * 128);  float* r0 = salloc((size_t)BN * 128);
        float* l1 = salloc((size_t)MV * 128);  float* r1 = salloc((size_t)MV * 128);
        float* ya = salloc((size_t)BN * 128);  float* tA = salloc((size_t)BN * 128);
        float* tB = salloc((size_t)BN * 128);  float* u  = salloc((size_t)MV * 128);
        float* da = salloc((size_t)BN * 128);  float* dv = salloc((size_t)MV * 64);
        k_build_xa1<<<BN, 128, 0, stream>>>(xa_in, acc_a, resg, te_a, xa1);
        k_build_xv1<<<BN, 64, 0, stream>>>(xvin_v3, acc_v, pos0, pos1, P + p_pos, xv1);
        run_stp(stream, P, stpA, xa1, xv1, l0, r0, l1, r1, ya, tA, tB, u, da, dv);
        k_add<<<(BN * 128) / 256, 256, 0, stream>>>(xa_cur, xa_in, da, BN * 128);
        k_add<<<(MV * 64) / 256, 256, 0, stream>>>(xv_cur, xvin_v3, dv, MV * 64);
    }

    // ================= stage 3: messages (m0, m1, mb0, mb1) =================
    aoff = 0;
    {
        const size_t Me = (size_t)B_ * E_;
        float* A0 = salloc((size_t)BN * 128); float* V1 = salloc((size_t)MV * 128);
        float* enc = salloc(Me * 128);  float* rs = salloc(Me * 3);
        float* rad = salloc(Me * 128);  float* t1 = salloc(Me * 128);
        float* t2  = salloc(Me * 128);  float* psia = salloc(Me * 128);
        float* q   = salloc(Me * 128);  float* psiv = salloc(Me * 64);
        hipMemsetAsync(acc_a, 0, (size_t)BN * 128 * 4, stream);
        hipMemsetAsync(acc_v, 0, (size_t)MV * 64 * 4, stream);
        run_msg(stream, P, m0,  rij0, src0, dst0, E_,  xa_cur, xv_cur, acc_a, acc_v, A0, V1, enc, rs, rad, t1, t2, psia, q, psiv);
        run_msg(stream, P, m1,  rij1, src1, dst1, E_,  xa_cur, xv_cur, acc_a, acc_v, A0, V1, enc, rs, rad, t1, t2, psia, q, psiv);
        run_msg(stream, P, mb0, rb0,  srcb, dstb, EB_, xa_cur, xv_cur, acc_a, acc_v, A0, V1, enc, rs, rad, t1, t2, psia, q, psiv);
        run_msg(stream, P, mb1, rb1,  srcb, dstb, EB_, xa_cur, xv_cur, acc_a, acc_v, A0, V1, enc, rs, rad, t1, t2, psia, q, psiv);
        k_add<<<(BN * 128) / 256, 256, 0, stream>>>(xa_cur, xa_cur, acc_a, BN * 128);
        k_add<<<(MV * 64) / 256, 256, 0, stream>>>(xv_cur, xv_cur, acc_v, MV * 64);
    }

    // ================= stage 4: self_tens_prods B =================
    aoff = 0;
    {
        float* l0 = salloc((size_t)BN * 128);  float* r0 = salloc((size_t)BN * 128);
        float* l1 = salloc((size_t)MV * 128);  float* r1 = salloc((size_t)MV * 128);
        float* ya = salloc((size_t)BN * 128);  float* tA = salloc((size_t)BN * 128);
        float* tB = salloc((size_t)BN * 128);  float* u  = salloc((size_t)MV * 128);
        float* da = salloc((size_t)BN * 128);  float* dv = salloc((size_t)MV * 64);
        run_stp(stream, P, stpB, xa_cur, xv_cur, l0, r0, l1, r1, ya, tA, tB, u, da, dv);
        k_add<<<(BN * 128) / 256, 256, 0, stream>>>(xa_cur, xa_cur, da, BN * 128);
        k_add<<<(MV * 64) / 256, 256, 0, stream>>>(xv_cur, xv_cur, dv, MV * 64);
    }

    // ================= stage 5: time-conditioned linear modulation =================
    aoff = 0;
    {
        float* tA = salloc((size_t)BN * 128);
        float* vT = salloc((size_t)MV * 64);
        gemm(stream, xa_cur, P + p_ta_i, nullptr, tA, BN, 128, 128, 0);
        k_scale_bcast<<<(BN * 128) / 256, 256, 0, stream>>>(tA, tlmte_a, NN_, 128, BN * 128);
        gemm(stream, tA, P + p_ta_o, nullptr, xa_cur, BN, 128, 128, GF_ACC);
        gemm(stream, xv_cur, P + p_tv_i, nullptr, vT, MV, 64, 64, 0);
        k_scale_bcast<<<(MV * 64) / 256, 256, 0, stream>>>(vT, tlmte_v, NN_ * 3, 64, MV * 64);
        gemm(stream, vT, P + p_tv_o, nullptr, xv_cur, MV, 64, 64, GF_ACC);
    }

    // ================= stage 6: emit (x_a, x_v) =================
    k_out<<<BN, 128, 0, stream>>>(xa_cur, xv_cur, out);
}
